// Transformer_Block_38817914421871
// MI455X (gfx1250) — matrix operand order repaired and checked by reading, not compiled, not run on hardware
//
#include <hip/hip_runtime.h>
#include <hip/hip_bf16.h>

#define EMBED 768
#define NTOK  8192          // 8 * 1024 tokens
#define ROWS  98304         // NTOK * 12 rows of 64
#define D     64
#define SEQ   1024
#define NHB   96            // head blocks = 8 batches * 12 heads
#define LOG2E 1.44269504088896340736f

typedef __attribute__((ext_vector_type(16))) __bf16 v16bf;
typedef __attribute__((ext_vector_type(8)))  float  v8f;
typedef int v4i_g __attribute__((vector_size(16)));   // GCC-style <4 x i32>

union Frag {
    v16bf v;
    uint4 q[2];
    unsigned short u[16];
};

#if __has_builtin(__builtin_amdgcn_cvt_pk_bf16_f32)
typedef __attribute__((ext_vector_type(2))) __bf16 v2bf;
__device__ __forceinline__ unsigned int pack2(float a, float b) {
    v2bf r = __builtin_amdgcn_cvt_pk_bf16_f32(a, b);
    return __builtin_bit_cast(unsigned int, r);
}
#else
__device__ __forceinline__ unsigned short f2bf_sw(float x) {
    unsigned int u = __float_as_uint(x);
    u += 0x7fffu + ((u >> 16) & 1u);   // round to nearest even
    return (unsigned short)(u >> 16);
}
__device__ __forceinline__ unsigned int pack2(float a, float b) {
    return (unsigned int)f2bf_sw(a) | ((unsigned int)f2bf_sw(b) << 16);
}
#endif

__device__ __forceinline__ float exp2_fast(float x) {
#if __has_builtin(__builtin_amdgcn_exp2f)
    return __builtin_amdgcn_exp2f(x);
#else
    return exp2f(x);
#endif
}

// ---- async global->LDS staging (gfx1250 GLOBAL_LOAD_ASYNC_TO_LDS_B128) ----
#if __has_builtin(__builtin_amdgcn_global_load_async_to_lds_b128)
#define HAS_ASYNC 1
__device__ __forceinline__ void stage16(const unsigned short* g, unsigned short* l) {
    __builtin_amdgcn_global_load_async_to_lds_b128(
        (__attribute__((address_space(1))) v4i_g*)g,
        (__attribute__((address_space(3))) v4i_g*)l,
        0, 0);
}
#else
#define HAS_ASYNC 0
__device__ __forceinline__ void stage16(const unsigned short* g, unsigned short* l) {
    *reinterpret_cast<uint4*>(l) = *reinterpret_cast<const uint4*>(g);
}
#endif

__device__ __forceinline__ void wait_async_le2() {
#if HAS_ASYNC
#if __has_builtin(__builtin_amdgcn_s_wait_asynccnt)
    __builtin_amdgcn_s_wait_asynccnt(2);
#else
    asm volatile("s_wait_asynccnt 2" ::: "memory");
#endif
#endif
}
__device__ __forceinline__ void wait_async_0() {
#if HAS_ASYNC
#if __has_builtin(__builtin_amdgcn_s_wait_asynccnt)
    __builtin_amdgcn_s_wait_asynccnt(0);
#else
    asm volatile("s_wait_asynccnt 0" ::: "memory");
#endif
#endif
}

// ---------------------------------------------------------------------------
// Kernel 0: one-time conversion of the three 64x64 f32 weight matrices to
// bf16 (row-major, rows = output columns => contiguous B-fragment columns).
// ---------------------------------------------------------------------------
__global__ __launch_bounds__(256) void wconv_kernel(const float* __restrict__ Wq,
                                                    const float* __restrict__ Wk,
                                                    const float* __restrict__ Wv,
                                                    unsigned short* __restrict__ Wb) {
    const int t = blockIdx.x * 256 + threadIdx.x;   // 0..1535, 8 elements each
    const float* src = (t < 512) ? Wq : (t < 1024) ? Wk : Wv;
    const int off = (t & 511) * 8;
    const float4* p = reinterpret_cast<const float4*>(src + off);
    float4 a = p[0], b = p[1];
    uint4 o;
    o.x = pack2(a.x, a.y);
    o.y = pack2(a.z, a.w);
    o.z = pack2(b.x, b.y);
    o.w = pack2(b.z, b.w);
    *reinterpret_cast<uint4*>(Wb + (size_t)t * 8) = o;
}

// ---------------------------------------------------------------------------
// Kernel 1: LayerNorm over 768, output bf16. One wave per token; each lane
// owns 24 contiguous columns (b128 loads/stores).
// ---------------------------------------------------------------------------
__global__ __launch_bounds__(256) void ln_kernel(const float* __restrict__ x,
                                                 const float* __restrict__ lw,
                                                 const float* __restrict__ lb,
                                                 unsigned short* __restrict__ xn) {
    const int wave = threadIdx.x >> 5;
    const int lane = threadIdx.x & 31;
    const int tok  = blockIdx.x * 8 + wave;

    const float4* px = reinterpret_cast<const float4*>(x + (size_t)tok * EMBED + lane * 24);
    float vals[24];
    float s = 0.f, ss = 0.f;
#pragma unroll
    for (int i = 0; i < 6; ++i) {
        float4 t = px[i];
        vals[4*i+0] = t.x; vals[4*i+1] = t.y; vals[4*i+2] = t.z; vals[4*i+3] = t.w;
        s  += t.x + t.y + t.z + t.w;
        ss += t.x*t.x + t.y*t.y + t.z*t.z + t.w*t.w;
    }
#pragma unroll
    for (int m = 16; m >= 1; m >>= 1) {
        s  += __shfl_xor(s,  m, 32);
        ss += __shfl_xor(ss, m, 32);
    }
    const float mu   = s * (1.0f / EMBED);
    const float var  = ss * (1.0f / EMBED) - mu * mu;
    const float rstd = rsqrtf(var + 1e-5f);

    const float4* pw = reinterpret_cast<const float4*>(lw + lane * 24);
    const float4* pb = reinterpret_cast<const float4*>(lb + lane * 24);
    float y[24];
#pragma unroll
    for (int i = 0; i < 6; ++i) {
        float4 w4 = pw[i];
        float4 b4 = pb[i];
        y[4*i+0] = (vals[4*i+0] - mu) * rstd * w4.x + b4.x;
        y[4*i+1] = (vals[4*i+1] - mu) * rstd * w4.y + b4.y;
        y[4*i+2] = (vals[4*i+2] - mu) * rstd * w4.z + b4.z;
        y[4*i+3] = (vals[4*i+3] - mu) * rstd * w4.w + b4.w;
    }
    uint4* dst = reinterpret_cast<uint4*>(xn + (size_t)tok * EMBED + lane * 24);
#pragma unroll
    for (int i = 0; i < 3; ++i) {
        uint4 o;
        o.x = pack2(y[8*i+0], y[8*i+1]);
        o.y = pack2(y[8*i+2], y[8*i+3]);
        o.z = pack2(y[8*i+4], y[8*i+5]);
        o.w = pack2(y[8*i+6], y[8*i+7]);
        dst[i] = o;
    }
}

// ---------------------------------------------------------------------------
// Kernel 2: QKV projection GEMM (98304x64) @ (64x64)^T + bias, via WMMA.
// Q additionally pre-scaled by log2(e) so attention softmax can use raw exp2.
// Q,K stored row-major bf16; V stored transposed per head-block.
// ---------------------------------------------------------------------------
__global__ __launch_bounds__(256) void qkv_kernel(const unsigned short* __restrict__ xn,
                                                  const unsigned short* __restrict__ Wb,
                                                  const float* __restrict__ bq,
                                                  const float* __restrict__ bk,
                                                  const float* __restrict__ bv,
                                                  unsigned short* __restrict__ Q,
                                                  unsigned short* __restrict__ K,
                                                  unsigned short* __restrict__ Vt) {
    const int wave = threadIdx.x >> 5;
    const int lane = threadIdx.x & 31;
    const int tile = blockIdx.x * 8 + wave;   // 0..6143
    const int base = tile * 16;               // global row base
    const int n = lane & 15;
    const int c = lane >> 4;

    Frag a0, a1;
    {
        const uint4* pr = reinterpret_cast<const uint4*>(xn + (size_t)(base + n) * D);
        a0.q[0] = pr[2 * c];     a0.q[1] = pr[2 * c + 1];
        a1.q[0] = pr[4 + 2 * c]; a1.q[1] = pr[5 + 2 * c];
    }

    const float* Bs[3] = {bq, bk, bv};

#pragma unroll
    for (int mat = 0; mat < 3; ++mat) {
        const unsigned short* W = Wb + (size_t)mat * 4096;
        const float* B = Bs[mat];
        const float scale = (mat == 0) ? LOG2E : 1.0f;
        v8f acc[4];
#pragma unroll
        for (int nt = 0; nt < 4; ++nt) {
            acc[nt] = v8f{};
#pragma unroll
            for (int s = 0; s < 2; ++s) {
                Frag bf;
                const uint4* w4 = reinterpret_cast<const uint4*>(W + (size_t)(nt * 16 + n) * D + s * 32 + c * 16);
                bf.q[0] = w4[0];
                bf.q[1] = w4[1];
                acc[nt] = __builtin_amdgcn_wmma_f32_16x16x32_bf16(
                    false, (s == 0) ? a0.v : a1.v, false, bf.v, (short)0, acc[nt], false, false);
            }
            const float bias = B[nt * 16 + n];
#pragma unroll
            for (int r = 0; r < 8; ++r) acc[nt][r] = (acc[nt][r] + bias) * scale;
        }

        if (mat < 2) {
            unsigned short* dst = (mat == 0) ? Q : K;
#pragma unroll
            for (int nt = 0; nt < 4; ++nt) {
                const int col = nt * 16 + n;
#pragma unroll
                for (int j = 0; j < 4; ++j) {
                    const unsigned int pr = pack2(acc[nt][2*j], acc[nt][2*j+1]);
                    dst[(size_t)(base + 2*j + 0 + 8*c) * D + col] = (unsigned short)pr;
                    dst[(size_t)(base + 2*j + 1 + 8*c) * D + col] = (unsigned short)(pr >> 16);
                }
            }
        } else {
            const int hb = base / SEQ;
            const int kl = (base % SEQ) + 8 * c;
#pragma unroll
            for (int nt = 0; nt < 4; ++nt) {
                uint4 o;
                o.x = pack2(acc[nt][0], acc[nt][1]);
                o.y = pack2(acc[nt][2], acc[nt][3]);
                o.z = pack2(acc[nt][4], acc[nt][5]);
                o.w = pack2(acc[nt][6], acc[nt][7]);
                *reinterpret_cast<uint4*>(Vt + (size_t)hb * (D * SEQ) + (size_t)(nt * 16 + n) * SEQ + kl) = o;
            }
        }
    }
}

// ---------------------------------------------------------------------------
// Kernel 3: flash attention. 8 waves/block = 8 query tiles of one head.
// K/V 32-key tiles double-buffered in LDS via async global->LDS loads.
// ---------------------------------------------------------------------------
__global__ __launch_bounds__(256) void attn_kernel(const unsigned short* __restrict__ Q,
                                                   const unsigned short* __restrict__ K,
                                                   const unsigned short* __restrict__ Vt,
                                                   float* __restrict__ out) {
    __shared__ __align__(16) unsigned char  KVbuf[2][8192];   // [0..4095]=K tile, [4096..8191]=V tile
    __shared__ __align__(16) unsigned short Plds[8][512];

    const int tid  = threadIdx.x;
    const int wave = tid >> 5;
    const int lane = tid & 31;
    const int n = lane & 15;
    const int c = lane >> 4;
    const int hb = blockIdx.x >> 3;   // head block 0..95
    const int qg = blockIdx.x & 7;    // query group 0..7
    const size_t qbase = (size_t)hb * SEQ + (qg * 8 + wave) * 16;

    const unsigned short* kblk = K  + (size_t)hb * SEQ * D;
    const unsigned short* vblk = Vt + (size_t)hb * (D * SEQ);

    // staging geometry: K tile 32 rows x 128B, V tile 64 rows x 64B
    const int krow = tid >> 3, kcc = tid & 7;
    const int vrow = tid >> 2, vcc = tid & 3;

    // prologue: stage key-tile 0 into buffer 0
    stage16(kblk + (size_t)krow * D + kcc * 8,      (unsigned short*)&KVbuf[0][tid * 16]);
    stage16(vblk + (size_t)vrow * SEQ + vcc * 8,    (unsigned short*)&KVbuf[0][4096 + tid * 16]);

    // Q A-fragments (16 queries x 64 dims => two K=32 fragments)
    Frag qa0, qa1;
    {
        const uint4* qp = reinterpret_cast<const uint4*>(Q + (qbase + n) * D);
        qa0.q[0] = qp[2 * c];     qa0.q[1] = qp[2 * c + 1];
        qa1.q[0] = qp[4 + 2 * c]; qa1.q[1] = qp[5 + 2 * c];
    }

    v8f o0 = v8f{}, o1 = v8f{}, o2 = v8f{}, o3 = v8f{};
    float mrow[8], lrow[8];                 // lrow = per-lane PARTIAL sums
#pragma unroll
    for (int r = 0; r < 8; ++r) { mrow[r] = -1e30f; lrow[r] = 0.f; }

    unsigned short* Pw = Plds[wave];
    int buf = 0;

    for (int kt = 0; kt < SEQ / 32; ++kt) {
        if (kt + 1 < SEQ / 32) {
            const size_t nk = (size_t)(kt + 1) * 32;
            stage16(kblk + (size_t)(nk + krow) * D + kcc * 8, (unsigned short*)&KVbuf[buf ^ 1][tid * 16]);
            stage16(vblk + (size_t)vrow * SEQ + nk + vcc * 8, (unsigned short*)&KVbuf[buf ^ 1][4096 + tid * 16]);
            wait_async_le2();   // previous tile's 2 loads complete (in-order), next 2 in flight
        } else {
            wait_async_0();
        }
        __syncthreads();

        const unsigned short* kb = (const unsigned short*)&KVbuf[buf][0];
        const unsigned short* vb = (const unsigned short*)&KVbuf[buf][4096];

        // ---- S = Q K^T (log2-scaled since Q carries log2e)
        v8f s0 = v8f{}, s1 = v8f{};
        {
            const uint4* kpA = reinterpret_cast<const uint4*>(kb + (size_t)n * D);
            const uint4* kpB = reinterpret_cast<const uint4*>(kb + (size_t)(16 + n) * D);
            Frag b;
            b.q[0] = kpA[2 * c];     b.q[1] = kpA[2 * c + 1];
            s0 = __builtin_amdgcn_wmma_f32_16x16x32_bf16(false, qa0.v, false, b.v, (short)0, s0, false, false);
            b.q[0] = kpA[4 + 2 * c]; b.q[1] = kpA[4 + 2 * c + 1];
            s0 = __builtin_amdgcn_wmma_f32_16x16x32_bf16(false, qa1.v, false, b.v, (short)0, s0, false, false);
            b.q[0] = kpB[2 * c];     b.q[1] = kpB[2 * c + 1];
            s1 = __builtin_amdgcn_wmma_f32_16x16x32_bf16(false, qa0.v, false, b.v, (short)0, s1, false, false);
            b.q[0] = kpB[4 + 2 * c]; b.q[1] = kpB[4 + 2 * c + 1];
            s1 = __builtin_amdgcn_wmma_f32_16x16x32_bf16(false, qa1.v, false, b.v, (short)0, s1, false, false);
        }

        // ---- online softmax in exp2 domain
        float alpha[8];
#pragma unroll
        for (int r = 0; r < 8; ++r) {
            float mx = fmaxf(s0[r], s1[r]);
#pragma unroll
            for (int msk = 8; msk >= 1; msk >>= 1) mx = fmaxf(mx, __shfl_xor(mx, msk, 16));
            const float mn = fmaxf(mrow[r], mx);
            alpha[r] = exp2_fast(mrow[r] - mn);
            const float p0 = exp2_fast(s0[r] - mn);
            const float p1 = exp2_fast(s1[r] - mn);
            lrow[r] = lrow[r] * alpha[r] + (p0 + p1);
            mrow[r] = mn;
            const int row = r + 8 * c;
            const unsigned int pp = pack2(p0, p1);
            Pw[row * 32 + n]      = (unsigned short)pp;
            Pw[row * 32 + 16 + n] = (unsigned short)(pp >> 16);
        }
#pragma unroll
        for (int r = 0; r < 8; ++r) {
            o0[r] *= alpha[r]; o1[r] *= alpha[r]; o2[r] *= alpha[r]; o3[r] *= alpha[r];
        }

        asm volatile("s_wait_dscnt 0" ::: "memory");

        // ---- P (16x32) in A layout from this wave's LDS staging tile
        Frag pa;
        {
            const uint4* pp = reinterpret_cast<const uint4*>(Pw + n * 32);
        pa.q[0] = pp[2 * c];
        pa.q[1] = pp[2 * c + 1];
        }

        // ---- O += P V : V tile rows (dims) are contiguous 32 keys in LDS
#pragma unroll
        for (int dt = 0; dt < 4; ++dt) {
            Frag vbf;
            const uint4* vp = reinterpret_cast<const uint4*>(vb + (size_t)(dt * 16 + n) * 32 + c * 16);
            vbf.q[0] = vp[0]; vbf.q[1] = vp[1];
            v8f& od = (dt == 0) ? o0 : (dt == 1) ? o1 : (dt == 2) ? o2 : o3;
            od = __builtin_amdgcn_wmma_f32_16x16x32_bf16(false, pa.v, false, vbf.v, (short)0, od, false, false);
        }

        asm volatile("s_wait_dscnt 0" ::: "memory");
        __syncthreads();   // all waves done reading buf before it is overwritten
        buf ^= 1;
    }

    // ---- finish deferred row-sum reduction, then normalize: 1/(8*l)
#pragma unroll
    for (int r = 0; r < 8; ++r) {
        float l = lrow[r];
#pragma unroll
        for (int msk = 8; msk >= 1; msk >>= 1) l += __shfl_xor(l, msk, 16);
        lrow[r] = l;
    }
#pragma unroll
    for (int r = 0; r < 8; ++r) {
        const float inv = 1.0f / (8.0f * lrow[r]);
        const size_t row = (qbase + r + 8 * c) * D;
        out[row + 0  + n] = o0[r] * inv;
        out[row + 16 + n] = o1[r] * inv;
        out[row + 32 + n] = o2[r] * inv;
        out[row + 48 + n] = o3[r] * inv;
    }
}

// ---------------------------------------------------------------------------
extern "C" void kernel_launch(void* const* d_in, const int* in_sizes, int n_in,
                              void* d_out, int out_size, void* d_ws, size_t ws_size,
                              hipStream_t stream) {
    const float* x    = (const float*)d_in[0];
    const float* ln_w = (const float*)d_in[1];
    const float* ln_b = (const float*)d_in[2];
    const float* Wq   = (const float*)d_in[3];
    const float* bq   = (const float*)d_in[4];
    const float* Wk   = (const float*)d_in[5];
    const float* bk   = (const float*)d_in[6];
    const float* Wv   = (const float*)d_in[7];
    const float* bv   = (const float*)d_in[8];
    float* out = (float*)d_out;

    unsigned short* xn = (unsigned short*)d_ws;
    unsigned short* Q  = xn + (size_t)ROWS * D;
    unsigned short* K  = Q  + (size_t)ROWS * D;
    unsigned short* Vt = K  + (size_t)ROWS * D;
    unsigned short* Wb = Vt + (size_t)ROWS * D;   // 3 * 64 * 64 bf16

    wconv_kernel<<<6, 256, 0, stream>>>(Wq, Wk, Wv, Wb);
    ln_kernel<<<NTOK / 8, 256, 0, stream>>>(x, ln_w, ln_b, xn);
    qkv_kernel<<<(ROWS / 16) / 8, 256, 0, stream>>>(xn, Wb, bq, bk, bv, Q, K, Vt);
    attn_kernel<<<NHB * 8, 256, 0, stream>>>(Q, K, Vt, out);
}
